// GemmaCausalSelfAttention_8108898255496
// MI455X (gfx1250) — matrix operand order repaired and checked by reading, not compiled, not run on hardware
//
#include <hip/hip_runtime.h>
#include <hip/hip_bf16.h>
#include <math.h>

typedef __bf16 bf16;
typedef __attribute__((ext_vector_type(16))) __bf16 v16bf;
typedef __attribute__((ext_vector_type(8)))  __bf16 v8bf;
typedef __attribute__((ext_vector_type(8)))  float  v8f;

#define B_   2
#define T_   2048
#define C_   2048
#define NH_  16
#define HD_  128

// ---------------------------------------------------------------------------
// WMMA fragment loaders (bf16, 16x16x32)
//
// A (16x32, MxK): lane L holds row m = L&15. Vector elems 0..7 -> K =
//   (L<16?0:8)+k0..  ; elems 8..15 -> K = 16+(L<16?0:8)+k0..  (ISA 7.12.2)
// B (32x16, KxN): lane L holds column n = L&15; elems 0..7 -> K =
//   (L<16?0:8)+k0.. ; elems 8..15 -> K = 16+(L<16?0:8)+k0.. (same as A)
// C/D (16x16 f32): lane L holds column n = L&15, VGPR v -> row m = v+8*(L>=16)
// ---------------------------------------------------------------------------
__device__ __forceinline__ v16bf load_afrag(const bf16* __restrict__ row, int k0, int lane) {
    const int half = lane >> 4;
    v8bf lo = *(const v8bf*)(row + k0 + half * 8);
    v8bf hi = *(const v8bf*)(row + k0 + 16 + half * 8);
    v16bf r;
#pragma unroll
    for (int i = 0; i < 8; ++i) { r[i] = lo[i]; r[8 + i] = hi[i]; }
    return r;
}

__device__ __forceinline__ v16bf load_bfrag(const bf16* __restrict__ col, int k0, int lane) {
    const int half = lane >> 4;
    v8bf lo = *(const v8bf*)(col + k0 + half * 8);
    v8bf hi = *(const v8bf*)(col + k0 + 16 + half * 8);
    v16bf r;
#pragma unroll
    for (int i = 0; i < 8; ++i) { r[i] = lo[i]; r[8 + i] = hi[i]; }
    return r;
}

#define WMMA_BF16(a, b, c) \
    __builtin_amdgcn_wmma_f32_16x16x32_bf16(false, (a), false, (b), (short)0, (c), false, false)

// ---------------------------------------------------------------------------
// Elementwise prep kernels
// ---------------------------------------------------------------------------
__global__ void cvt_f32_bf16(const float* __restrict__ in, bf16* __restrict__ out, long long n) {
    long long i = (long long)blockIdx.x * blockDim.x + threadIdx.x;
    if (i < n) out[i] = (bf16)in[i];
}

// in: K x N row-major f32 -> out: N x K row-major bf16
__global__ void transpose_cvt(const float* __restrict__ in, bf16* __restrict__ out, int K, int N) {
    long long i = (long long)blockIdx.x * blockDim.x + threadIdx.x;
    if (i < (long long)K * N) {
        int k = (int)(i / N), n = (int)(i % N);
        out[(size_t)n * K + k] = (bf16)in[i];
    }
}

// RoPE + f32->bf16, generic over number of heads (row stride = nh*HD_)
__global__ void rope_cvt(const float* __restrict__ in, bf16* __restrict__ out,
                         const float* __restrict__ cosb, const float* __restrict__ sinb, int nh) {
    long long i = (long long)blockIdx.x * blockDim.x + threadIdx.x;
    long long total = (long long)B_ * T_ * nh * (HD_ / 2);
    if (i >= total) return;
    int p = (int)(i & 63);
    long long j = i >> 6;
    int h = (int)(j % nh); j /= nh;
    int t = (int)(j % T_);
    int b = (int)(j / T_);
    size_t base = ((size_t)(b * T_ + t)) * ((size_t)nh * HD_) + (size_t)h * HD_ + 2 * p;
    float re = in[base], im = in[base + 1];
    float c = cosb[t * 64 + p], s = sinb[t * 64 + p];
    out[base]     = (bf16)(re * c - im * s);
    out[base + 1] = (bf16)(re * s + im * c);
}

// v [B*T, 128] f32 -> vT [B, 128, T] bf16 (so P@V B-fragments are contiguous)
__global__ void v_transpose_cvt(const float* __restrict__ in, bf16* __restrict__ out) {
    long long i = (long long)blockIdx.x * blockDim.x + threadIdx.x;
    if (i >= (long long)B_ * T_ * HD_) return;
    int d = (int)(i % HD_);
    long long j = i / HD_;
    int t = (int)(j % T_);
    int b = (int)(j / T_);
    out[((size_t)b * HD_ + d) * T_ + t] = (bf16)in[i];
}

// ---------------------------------------------------------------------------
// GEMM: C[M,N] = A[M,K](bf16) * Bt[N,K](bf16)^T + bias, f32 accumulate.
// One wave per block, 32x64 output tile. K = C_ is compile-time constant.
// K loop unrolled by 2 with alternating fragment buffers: zero rotation
// copies, loads for the next 32-K step always in flight over the current
// WMMAs (partial s_wait_loadcnt instead of a full drain).
// ---------------------------------------------------------------------------
__global__ void __launch_bounds__(32)
gemm_bf16_wmma(const bf16* __restrict__ A, const bf16* __restrict__ Bt,
               const float* __restrict__ bias, float* __restrict__ Cmat,
               int M, int N) {
    constexpr int K = C_;  // reduction dim is the model width for all GEMMs
    const int lane = threadIdx.x & 31;
    const int r = lane & 15;
    const int half = lane >> 4;
    const int m0 = blockIdx.y * 32;
    const int n0 = blockIdx.x * 64;

    const bf16* arow[2];
    const bf16* bcol[4];
#pragma unroll
    for (int i = 0; i < 2; ++i) arow[i] = A  + (size_t)(m0 + i * 16 + r) * K;
#pragma unroll
    for (int j = 0; j < 4; ++j) bcol[j] = Bt + (size_t)(n0 + j * 16 + r) * K;

    v8f acc[2][4] = {};

    v16bf a0[2], b0[4], a1[2], b1[4];
#pragma unroll
    for (int i = 0; i < 2; ++i) a0[i] = load_afrag(arow[i], 0, lane);
#pragma unroll
    for (int j = 0; j < 4; ++j) b0[j] = load_bfrag(bcol[j], 0, lane);

#pragma unroll 1
    for (int k0 = 0; k0 < K - 64; k0 += 64) {
        // L2 prefetch ~2KB ahead (gfx1250 global_prefetch path)
        __builtin_prefetch(arow[0] + k0 + 1024, 0, 3);
        __builtin_prefetch(bcol[0] + k0 + 1024, 0, 3);

        // fetch k0+32 into buffer 1, consume buffer 0
#pragma unroll
        for (int i = 0; i < 2; ++i) a1[i] = load_afrag(arow[i], k0 + 32, lane);
#pragma unroll
        for (int j = 0; j < 4; ++j) b1[j] = load_bfrag(bcol[j], k0 + 32, lane);
#pragma unroll
        for (int i = 0; i < 2; ++i)
#pragma unroll
            for (int j = 0; j < 4; ++j)
                acc[i][j] = WMMA_BF16(a0[i], b0[j], acc[i][j]);

        // fetch k0+64 into buffer 0, consume buffer 1
#pragma unroll
        for (int i = 0; i < 2; ++i) a0[i] = load_afrag(arow[i], k0 + 64, lane);
#pragma unroll
        for (int j = 0; j < 4; ++j) b0[j] = load_bfrag(bcol[j], k0 + 64, lane);
#pragma unroll
        for (int i = 0; i < 2; ++i)
#pragma unroll
            for (int j = 0; j < 4; ++j)
                acc[i][j] = WMMA_BF16(a1[i], b1[j], acc[i][j]);
    }

    // tail: last 64 K (buffer 0 holds K-64, fetch K-32 into buffer 1)
#pragma unroll
    for (int i = 0; i < 2; ++i) a1[i] = load_afrag(arow[i], K - 32, lane);
#pragma unroll
    for (int j = 0; j < 4; ++j) b1[j] = load_bfrag(bcol[j], K - 32, lane);
#pragma unroll
    for (int i = 0; i < 2; ++i)
#pragma unroll
        for (int j = 0; j < 4; ++j)
            acc[i][j] = WMMA_BF16(a0[i], b0[j], acc[i][j]);
#pragma unroll
    for (int i = 0; i < 2; ++i)
#pragma unroll
        for (int j = 0; j < 4; ++j)
            acc[i][j] = WMMA_BF16(a1[i], b1[j], acc[i][j]);

#pragma unroll
    for (int i = 0; i < 2; ++i) {
#pragma unroll
        for (int j = 0; j < 4; ++j) {
            int n = n0 + j * 16 + r;
            float bb = bias ? bias[n] : 0.0f;
#pragma unroll
            for (int v = 0; v < 8; ++v)
                Cmat[(size_t)(m0 + i * 16 + v + 8 * half) * N + n] = acc[i][j][v] + bb;
        }
    }
}

// ---------------------------------------------------------------------------
// Flash attention: one wave per (b, h, 16-query tile). Streams 32 keys/step.
// qb: [B*T, C] bf16 (rope'd), kb: [B*T, 128] bf16 (rope'd), vT: [B,128,T] bf16
// out ab: [B*T, C] bf16 (softmax(QK^T/sqrt(d)+causal) @ V)
// ---------------------------------------------------------------------------
__global__ void __launch_bounds__(32)
flash_attn(const bf16* __restrict__ qb, const bf16* __restrict__ kb,
           const bf16* __restrict__ vT, bf16* __restrict__ ab) {
    __shared__ bf16 plds[16][40];   // 16x32 P tile, padded rows (80B stride)

    const int lane = threadIdx.x & 31;
    const int r = lane & 15;
    const int half = lane >> 4;
    const int q0 = blockIdx.x * 16;
    const int h = blockIdx.y % NH_;
    const int b = blockIdx.y / NH_;

    // Q fragments (16 rows x 128 dims) held in registers for the whole pass
    const bf16* qrow = qb + ((size_t)(b * T_ + q0 + r)) * C_ + (size_t)h * HD_;
    v16bf aq[4];
#pragma unroll
    for (int kc = 0; kc < 4; ++kc) aq[kc] = load_afrag(qrow, kc * 32, lane);

    v8f O[8] = {};
    float rmax[8], rsum[8];
#pragma unroll
    for (int v = 0; v < 8; ++v) { rmax[v] = -3.0e38f; rsum[v] = 0.0f; }

    const float scale = 0.08838834764831845f;  // 1/sqrt(128)

    for (int n0 = 0; n0 < q0 + 16; n0 += 32) {
        // ---- S = Q K^T for 32 keys (two 16-key column tiles) ----
        v8f S0 = {}, S1 = {};
        const bf16* krow0 = kb + ((size_t)(b * T_ + n0 + r)) * HD_;
        const bf16* krow1 = krow0 + (size_t)16 * HD_;
#pragma unroll
        for (int kc = 0; kc < 4; ++kc) {
            v16bf b0 = load_bfrag(krow0, kc * 32, lane);
            v16bf b1 = load_bfrag(krow1, kc * 32, lane);
            S0 = WMMA_BF16(aq[kc], b0, S0);
            S1 = WMMA_BF16(aq[kc], b1, S1);
        }

        // ---- online softmax (rows of S live across 16 lanes, same VGPR) ----
        float p0[8], p1[8], fac[8];
#pragma unroll
        for (int v = 0; v < 8; ++v) {
            const int m = q0 + v + 8 * half;
            float s0 = S0[v] * scale; if (n0 + r      > m) s0 = -1.0e9f;
            float s1 = S1[v] * scale; if (n0 + 16 + r > m) s1 = -1.0e9f;
            float tm = fmaxf(s0, s1);
#pragma unroll
            for (int off = 1; off < 16; off <<= 1) tm = fmaxf(tm, __shfl_xor(tm, off, 32));
            float nm = fmaxf(rmax[v], tm);
            float f  = __expf(rmax[v] - nm);
            float e0 = __expf(s0 - nm);
            float e1 = __expf(s1 - nm);
            float ts = e0 + e1;
#pragma unroll
            for (int off = 1; off < 16; off <<= 1) ts += __shfl_xor(ts, off, 32);
            rsum[v] = rsum[v] * f + ts;
            rmax[v] = nm;
            fac[v] = f; p0[v] = e0; p1[v] = e1;
        }
#pragma unroll
        for (int j = 0; j < 8; ++j)
#pragma unroll
            for (int v = 0; v < 8; ++v) O[j][v] *= fac[v];

        // ---- C-layout -> A-layout for P via LDS ----
#pragma unroll
        for (int v = 0; v < 8; ++v) {
            plds[v + 8 * half][r]      = (bf16)p0[v];
            plds[v + 8 * half][r + 16] = (bf16)p1[v];
        }
        __syncthreads();
        v16bf ap = load_afrag(&plds[r][0], 0, lane);

        // ---- O += P @ V (8 x 16-wide head-dim chunks) ----
#pragma unroll
        for (int j = 0; j < 8; ++j) {
            const bf16* vrow = vT + ((size_t)b * HD_ + j * 16 + r) * T_;
            v16bf bv = load_bfrag(vrow, n0, lane);
            O[j] = WMMA_BF16(ap, bv, O[j]);
        }
        __syncthreads();
    }

    // ---- normalize + store bf16 for the output projection ----
#pragma unroll
    for (int j = 0; j < 8; ++j) {
        int n = h * HD_ + j * 16 + r;
#pragma unroll
        for (int v = 0; v < 8; ++v) {
            float o = O[j][v] / rsum[v];
            ab[((size_t)(b * T_ + q0 + v + 8 * half)) * C_ + n] = (bf16)o;
        }
    }
}

// ---------------------------------------------------------------------------
extern "C" void kernel_launch(void* const* d_in, const int* in_sizes, int n_in,
                              void* d_out, int out_size, void* d_ws, size_t ws_size,
                              hipStream_t stream) {
    const float* x    = (const float*)d_in[0];
    const float* fcos = (const float*)d_in[1];
    const float* fsin = (const float*)d_in[2];
    // d_in[3] = mask: causal mask recomputed inline
    const float* wq   = (const float*)d_in[4];
    const float* wq_b = (const float*)d_in[5];
    const float* wk   = (const float*)d_in[6];
    const float* wk_b = (const float*)d_in[7];
    const float* wv   = (const float*)d_in[8];
    const float* wv_b = (const float*)d_in[9];
    const float* wo   = (const float*)d_in[10];
    const float* wo_b = (const float*)d_in[11];
    float* out = (float*)d_out;

    const size_t M  = (size_t)B_ * T_;       // 4096 rows
    char* ws = (char*)d_ws;
    auto alloc = [&](size_t bytes) { void* p = (void*)ws; ws += (bytes + 255) & ~(size_t)255; return p; };

    bf16*  xb  = (bf16*)alloc(M * C_ * 2);
    bf16*  wqT = (bf16*)alloc((size_t)C_ * C_ * 2);
    bf16*  wkT = (bf16*)alloc((size_t)HD_ * C_ * 2);
    bf16*  wvT = (bf16*)alloc((size_t)HD_ * C_ * 2);
    bf16*  woT = (bf16*)alloc((size_t)C_ * C_ * 2);
    float* qf  = (float*)alloc(M * C_ * 4);
    float* kf  = (float*)alloc(M * HD_ * 4);
    float* vf  = (float*)alloc(M * HD_ * 4);
    bf16*  qb  = (bf16*)alloc(M * C_ * 2);
    bf16*  kb  = (bf16*)alloc(M * HD_ * 2);
    bf16*  vbT = (bf16*)alloc(M * HD_ * 2);
    bf16*  ab  = (bf16*)alloc(M * C_ * 2);

    const int TPB = 256;
    auto nblk = [&](long long n) { return (unsigned)((n + TPB - 1) / TPB); };

    // 1) convert activations / weights to bf16 (weights transposed to NxK)
    cvt_f32_bf16<<<nblk((long long)M * C_), TPB, 0, stream>>>(x, xb, (long long)M * C_);
    transpose_cvt<<<nblk((long long)C_ * C_), TPB, 0, stream>>>(wq, wqT, C_, C_);
    transpose_cvt<<<nblk((long long)C_ * HD_), TPB, 0, stream>>>(wk, wkT, C_, HD_);
    transpose_cvt<<<nblk((long long)C_ * HD_), TPB, 0, stream>>>(wv, wvT, C_, HD_);
    transpose_cvt<<<nblk((long long)C_ * C_), TPB, 0, stream>>>(wo, woT, C_, C_);

    // 2) projections (WMMA, 32x64 wave tiles)
    gemm_bf16_wmma<<<dim3(C_ / 64,  (unsigned)(M / 32)), 32, 0, stream>>>(xb, wqT, wq_b, qf, (int)M, C_);
    gemm_bf16_wmma<<<dim3(HD_ / 64, (unsigned)(M / 32)), 32, 0, stream>>>(xb, wkT, wk_b, kf, (int)M, HD_);
    gemm_bf16_wmma<<<dim3(HD_ / 64, (unsigned)(M / 32)), 32, 0, stream>>>(xb, wvT, wv_b, vf, (int)M, HD_);

    // 3) RoPE (+bf16 cvt) on q and k; transpose-convert v
    rope_cvt<<<nblk((long long)B_ * T_ * NH_ * 64), TPB, 0, stream>>>(qf, qb, fcos, fsin, NH_);
    rope_cvt<<<nblk((long long)B_ * T_ * 64),       TPB, 0, stream>>>(kf, kb, fcos, fsin, 1);
    v_transpose_cvt<<<nblk((long long)M * HD_),     TPB, 0, stream>>>(vf, vbT);

    // 4) causal flash attention (WMMA)
    flash_attn<<<dim3(T_ / 16, B_ * NH_), 32, 0, stream>>>(qb, kb, vbT, ab);

    // 5) output projection (WMMA) -> f32 result
    gemm_bf16_wmma<<<dim3(C_ / 64, (unsigned)(M / 32)), 32, 0, stream>>>(ab, woT, wo_b, out, (int)M, C_);
}
